// MultiHeadAttention_63393717289722
// MI455X (gfx1250) — matrix operand order repaired and checked by reading, not compiled, not run on hardware
//
#include <hip/hip_runtime.h>

// ---------------------------------------------------------------------------
// MHA forward for MI455X (gfx1250, wave32).
//  - all matmuls via v_wmma_f32_16x16x32_bf16 (bf16 operands, f32 accum)
//  - projection GEMMs stage their weight slab into LDS with
//    global_load_async_to_lds_b128 (ASYNCcnt) and serve B-fragments from LDS
//  - flash attention with online softmax, K read direct from global,
//    V transposed through LDS, P restaged through LDS (C->A layout hop)
// ---------------------------------------------------------------------------

typedef __bf16 bf16_t;
typedef __bf16 v16bf __attribute__((ext_vector_type(16)));
typedef __bf16 v8bf  __attribute__((ext_vector_type(8)));
typedef float  v8f   __attribute__((ext_vector_type(8)));

#define B_  4
#define S_  2048
#define D_  512
#define H_  8
#define DK_ 64

// ---- helpers --------------------------------------------------------------

__device__ static inline bf16_t f2bf(float x) {
  // native RNE f32 -> bf16 conversion (v_cvt on gfx1250)
  return (bf16_t)x;
}

__device__ static inline v8f wmma_bf16(v16bf a, v16bf b, v8f c) {
  // (neg_a, A, neg_b, B, c_mod, C, reuse_a, reuse_b)
  return __builtin_amdgcn_wmma_f32_16x16x32_bf16(false, a, false, b,
                                                 (short)0, c, false, false);
}

// Low 32 bits of a generic pointer to __shared__ == workgroup LDS offset.
__device__ static inline unsigned lds_off(const void* p) {
  return (unsigned)(uintptr_t)p;
}

// Async-copy a contiguous 64KB slab (64 rows x 512 bf16) global -> LDS.
// 128 threads x 32 iterations x 16B, tracked on ASYNCcnt.
__device__ static inline void async_copy_slab_64k(const bf16_t* __restrict__ src,
                                                  bf16_t* dstLds, int tid) {
  unsigned long long gb = (unsigned long long)(uintptr_t)src;
  unsigned lbase = lds_off(dstLds);
#pragma unroll 4
  for (int it = 0; it < 32; ++it) {
    unsigned off  = (unsigned)(tid + it * 128) * 16u;
    unsigned ldsa = lbase + off;
    // GVS mode: mem_addr = SADDR64 + VADDR32; VDST = LDS byte address
    asm volatile("global_load_async_to_lds_b128 %0, %1, %2"
                 :
                 : "v"(ldsa), "v"(off), "s"(gb)
                 : "memory");
  }
  asm volatile("s_wait_asynccnt 0" ::: "memory");
}

// A-fragment (16x32 bf16) from row-major [m][k] storage, leading dim ld.
// ISA layout: lane holds row (lane&15); kbase=(lane>>4)*8; elements
// i<8 -> k=kbase+i, i>=8 -> k=kbase+16+(i-8).
__device__ static inline v16bf load_a_bf16(const bf16_t* A, int ld, int lane, int k0) {
  const bf16_t* p = A + (size_t)(lane & 15) * ld + k0 + ((lane >> 4) << 3);
  v8bf lo = *(const v8bf*)p;
  v8bf hi = *(const v8bf*)(p + 16);
  v16bf a;
#pragma unroll
  for (int i = 0; i < 8; ++i) { a[i] = lo[i]; a[8 + i] = hi[i]; }
  return a;
}

// Same A-fragment but sourced from f32 and converted in-register.
__device__ static inline v16bf load_a_f32cvt(const float* A, int ld, int lane, int k0) {
  const float* p = A + (size_t)(lane & 15) * ld + k0 + ((lane >> 4) << 3);
  float4 x0 = *(const float4*)(p);
  float4 x1 = *(const float4*)(p + 4);
  float4 y0 = *(const float4*)(p + 16);
  float4 y1 = *(const float4*)(p + 20);
  v16bf a;
  a[0] = f2bf(x0.x);  a[1] = f2bf(x0.y);  a[2]  = f2bf(x0.z);  a[3]  = f2bf(x0.w);
  a[4] = f2bf(x1.x);  a[5] = f2bf(x1.y);  a[6]  = f2bf(x1.z);  a[7]  = f2bf(x1.w);
  a[8] = f2bf(y0.x);  a[9] = f2bf(y0.y);  a[10] = f2bf(y0.z);  a[11] = f2bf(y0.w);
  a[12] = f2bf(y1.x); a[13] = f2bf(y1.y); a[14] = f2bf(y1.z); a[15] = f2bf(y1.w);
  return a;
}

// B-fragment (32x16 bf16) from "n-major" storage Bn[n][k], leading dim ld.
// ISA layout: lane column n = n0+(lane&15); kbase=(lane>>4)*8; elements
// i<8 -> k=kbase+i, i>=8 -> k=kbase+16+(i-8) (same order as A).
__device__ static inline v16bf load_b_nk(const bf16_t* Bn, int ld, int lane,
                                         int n0, int k0) {
  const bf16_t* p = Bn + (size_t)(n0 + (lane & 15)) * ld + k0 + ((lane >> 4) << 3);
  v8bf lo = *(const v8bf*)p;
  v8bf hi = *(const v8bf*)(p + 16);
  v16bf b;
#pragma unroll
  for (int i = 0; i < 8; ++i) { b[i] = lo[i]; b[8 + i] = hi[i]; }
  return b;
}

// Reductions across the 16-lane groups that hold one C-fragment row.
__device__ static inline float redmax16(float v) {
#pragma unroll
  for (int m = 1; m < 16; m <<= 1) v = fmaxf(v, __shfl_xor(v, m, 32));
  return v;
}
__device__ static inline float redsum16(float v) {
#pragma unroll
  for (int m = 1; m < 16; m <<= 1) v += __shfl_xor(v, m, 32);
  return v;
}

// ---- kernels --------------------------------------------------------------

__global__ void cvt_f32_bf16(const float* __restrict__ src,
                             bf16_t* __restrict__ dst, int n) {
  int i = blockIdx.x * 256 + threadIdx.x;
  if (i < n) dst[i] = f2bf(src[i]);
}

// Y = X @ W.T, X f32 [8192,512], W bf16 [n][k] (native layout == B-fragment
// layout). The block's 64x512 W slab is async-staged into LDS once, then all
// B-fragments come from LDS. Output written as bf16 in [B,H,S,DK].
// grid (M/64, N/64), block 128 (4 waves, one 16x64 tile each).
__global__ __launch_bounds__(128) void proj_qkv(const float* __restrict__ X,
                                                const bf16_t* __restrict__ Wb,
                                                bf16_t* __restrict__ Outh) {
  __shared__ bf16_t Wl[64][D_];   // 64 KB weight slab for this n-block

  const int lane = threadIdx.x & 31;
  const int wid  = threadIdx.x >> 5;
  const int m0 = blockIdx.x * 64 + wid * 16;
  const int n0 = blockIdx.y * 64;
  const float* Am = X + (size_t)m0 * D_;

  async_copy_slab_64k(Wb + (size_t)n0 * D_, &Wl[0][0], threadIdx.x);
  __syncthreads();

  v8f acc[4] = {v8f{}, v8f{}, v8f{}, v8f{}};
  for (int k0 = 0; k0 < D_; k0 += 32) {
    v16bf a = load_a_f32cvt(Am, D_, lane, k0);
#pragma unroll
    for (int j = 0; j < 4; ++j)
      acc[j] = wmma_bf16(a, load_b_nk(&Wl[0][0], D_, lane, 16 * j, k0), acc[j]);
  }

  const int grp = lane >> 4, nn = lane & 15;
#pragma unroll
  for (int r = 0; r < 8; ++r) {
    int m = m0 + r + 8 * grp;
    int b = m >> 11, s = m & (S_ - 1);
#pragma unroll
    for (int j = 0; j < 4; ++j) {
      int n = n0 + 16 * j + nn;
      int h = n >> 6, dk = n & (DK_ - 1);
      Outh[(((size_t)b * H_ + h) * S_ + s) * DK_ + dk] = f2bf(acc[j][r]);
    }
  }
}

// Flash attention: one wave = one 16-row Q tile; block = 4 waves sharing
// one (b,h); stream over 32-key chunks with online softmax.
__global__ __launch_bounds__(128) void attn_fwd(const bf16_t* __restrict__ Qb,
                                                const bf16_t* __restrict__ Kb,
                                                const bf16_t* __restrict__ Vb,
                                                bf16_t* __restrict__ Ctx) {
  __shared__ bf16_t Vt[DK_][40];       // V chunk transposed [dk][key'], 80B rows
  __shared__ bf16_t Pst[4][16][32];    // per-wave P staging (C->A layout hop)

  const int lane = threadIdx.x & 31;
  const int wid  = threadIdx.x >> 5;
  const int bh = blockIdx.y;
  const int b = bh >> 3, h = bh & (H_ - 1);
  const int q0 = blockIdx.x * 64 + wid * 16;

  const bf16_t* Qh = Qb + (size_t)bh * S_ * DK_;
  const bf16_t* Kh = Kb + (size_t)bh * S_ * DK_;
  const bf16_t* Vh = Vb + (size_t)bh * S_ * DK_;

  // Q A-fragments (dk contraction split 0..31 / 32..63), kept in registers.
  v16bf aq0 = load_a_bf16(Qh + (size_t)q0 * DK_, DK_, lane, 0);
  v16bf aq1 = load_a_bf16(Qh + (size_t)q0 * DK_, DK_, lane, 32);

  v8f o0 = {}, o1 = {}, o2 = {}, o3 = {};
  float mrow[8], lrow[8];
#pragma unroll
  for (int r = 0; r < 8; ++r) { mrow[r] = -3.0e38f; lrow[r] = 0.0f; }
  const float scale = 0.125f;  // 1/sqrt(64)

  for (int c = 0; c < S_ / 32; ++c) {
    const int key0 = c * 32;
    __syncthreads();
    // Cooperatively transpose V chunk (32 keys x 64 dk) into LDS.
#pragma unroll
    for (int it = 0; it < 2; ++it) {
      int u  = threadIdx.x + it * 128;   // 0..255
      int kp = u >> 3;                   // key' 0..31
      int d0 = (u & 7) << 3;             // dk octet base
      v8bf vv = *(const v8bf*)(Vh + (size_t)(key0 + kp) * DK_ + d0);
#pragma unroll
      for (int j = 0; j < 8; ++j) Vt[d0 + j][kp] = vv[j];
    }
    __syncthreads();

    if (c + 1 < S_ / 32) {  // uniform branch; pull next chunk toward L2/L0
      __builtin_prefetch(Kh + (size_t)(key0 + 32) * DK_ + (lane << 4), 0, 1);
      __builtin_prefetch(Vh + (size_t)(key0 + 32) * DK_ + (lane << 4), 0, 1);
    }

    // S = Q K^T for 32 keys: two 16x16 score tiles, K read straight from
    // global ([key][dk] is already the n-major B layout).
    v8f s0 = {}, s1 = {};
    s0 = wmma_bf16(aq0, load_b_nk(Kh + (size_t)key0 * DK_, DK_, lane, 0, 0), s0);
    s0 = wmma_bf16(aq1, load_b_nk(Kh + (size_t)key0 * DK_, DK_, lane, 0, 32), s0);
    s1 = wmma_bf16(aq0, load_b_nk(Kh + (size_t)key0 * DK_, DK_, lane, 16, 0), s1);
    s1 = wmma_bf16(aq1, load_b_nk(Kh + (size_t)key0 * DK_, DK_, lane, 16, 32), s1);

    // Online softmax update (rows live at r + 8*(lane>>4), cols at lane&15).
    float alpha[8];
#pragma unroll
    for (int r = 0; r < 8; ++r) {
      float a0 = s0[r] * scale;
      float a1 = s1[r] * scale;
      float mx = redmax16(fmaxf(a0, a1));
      float mnew = fmaxf(mrow[r], mx);
      float al = __expf(mrow[r] - mnew);
      float p0 = __expf(a0 - mnew);
      float p1 = __expf(a1 - mnew);
      lrow[r] = lrow[r] * al + redsum16(p0 + p1);
      mrow[r] = mnew;
      alpha[r] = al;
      s0[r] = p0;
      s1[r] = p1;
    }
#pragma unroll
    for (int r = 0; r < 8; ++r) {
      o0[r] *= alpha[r]; o1[r] *= alpha[r];
      o2[r] *= alpha[r]; o3[r] *= alpha[r];
    }

    // Re-stage P through LDS: C-fragment layout -> A-fragment layout.
    // Same-wave LDS ops are in-order, so no barrier needed.
    const int grp = lane >> 4, nn = lane & 15;
#pragma unroll
    for (int r = 0; r < 8; ++r) {
      Pst[wid][r + 8 * grp][nn]      = f2bf(s0[r]);
      Pst[wid][r + 8 * grp][16 + nn] = f2bf(s1[r]);
    }
    v16bf pa = load_a_bf16(&Pst[wid][0][0], 32, lane, 0);

    // O += P V using the transposed V in LDS.
    o0 = wmma_bf16(pa, load_b_nk(&Vt[0][0], 40, lane, 0,  0), o0);
    o1 = wmma_bf16(pa, load_b_nk(&Vt[0][0], 40, lane, 16, 0), o1);
    o2 = wmma_bf16(pa, load_b_nk(&Vt[0][0], 40, lane, 32, 0), o2);
    o3 = wmma_bf16(pa, load_b_nk(&Vt[0][0], 40, lane, 48, 0), o3);
  }

  // Epilogue: normalize and write ctx as bf16 [B,S,D] (heads re-interleaved).
  const int grp = lane >> 4, nn = lane & 15;
#pragma unroll
  for (int r = 0; r < 8; ++r) {
    const int srow = q0 + r + 8 * grp;
    const float inv = 1.0f / lrow[r];
    bf16_t* dst = Ctx + ((size_t)b * S_ + srow) * D_ + h * DK_;
    dst[nn]      = f2bf(o0[r] * inv);
    dst[16 + nn] = f2bf(o1[r] * inv);
    dst[32 + nn] = f2bf(o2[r] * inv);
    dst[48 + nn] = f2bf(o3[r] * inv);
  }
}

// out = ctx @ Wo.T + bo, f32 output; Wo slab async-staged to LDS.
__global__ __launch_bounds__(128) void out_proj(const bf16_t* __restrict__ Ctx,
                                                const bf16_t* __restrict__ Wob,
                                                const float* __restrict__ bo,
                                                float* __restrict__ Out) {
  __shared__ bf16_t Wl[64][D_];   // 64 KB weight slab for this n-block

  const int lane = threadIdx.x & 31;
  const int wid  = threadIdx.x >> 5;
  const int m0 = blockIdx.x * 64 + wid * 16;
  const int n0 = blockIdx.y * 64;
  const bf16_t* Am = Ctx + (size_t)m0 * D_;

  async_copy_slab_64k(Wob + (size_t)n0 * D_, &Wl[0][0], threadIdx.x);
  __syncthreads();

  v8f acc[4] = {v8f{}, v8f{}, v8f{}, v8f{}};
  for (int k0 = 0; k0 < D_; k0 += 32) {
    v16bf a = load_a_bf16(Am, D_, lane, k0);
#pragma unroll
    for (int j = 0; j < 4; ++j)
      acc[j] = wmma_bf16(a, load_b_nk(&Wl[0][0], D_, lane, 16 * j, k0), acc[j]);
  }

  const int grp = lane >> 4, nn = lane & 15;
#pragma unroll
  for (int r = 0; r < 8; ++r) {
    int m = m0 + r + 8 * grp;
#pragma unroll
    for (int j = 0; j < 4; ++j) {
      int n = n0 + 16 * j + nn;
      Out[(size_t)m * D_ + n] = acc[j][r] + bo[n];
    }
  }
}

// ---- launch ---------------------------------------------------------------

extern "C" void kernel_launch(void* const* d_in, const int* in_sizes, int n_in,
                              void* d_out, int out_size, void* d_ws, size_t ws_size,
                              hipStream_t stream) {
  (void)in_sizes; (void)n_in; (void)out_size; (void)ws_size;

  const float* q  = (const float*)d_in[0];
  const float* k  = (const float*)d_in[1];
  const float* v  = (const float*)d_in[2];
  const float* Wq = (const float*)d_in[3];
  const float* Wk = (const float*)d_in[4];
  const float* Wv = (const float*)d_in[5];
  const float* Wo = (const float*)d_in[6];
  const float* bo = (const float*)d_in[7];
  float* out = (float*)d_out;

  // Workspace layout (bf16 elements): 4x head tensors + 4x weights = ~35.7 MB
  const size_t SZH = (size_t)B_ * H_ * S_ * DK_;   // 4,194,304
  bf16_t* qb   = (bf16_t*)d_ws;
  bf16_t* kb   = qb + SZH;
  bf16_t* vb   = kb + SZH;
  bf16_t* ctxb = vb + SZH;                          // [B,S,D], same count
  bf16_t* wqb  = ctxb + (size_t)B_ * S_ * D_;
  bf16_t* wkb  = wqb + (size_t)D_ * D_;
  bf16_t* wvb  = wkb + (size_t)D_ * D_;
  bf16_t* wob  = wvb + (size_t)D_ * D_;

  const int WN = D_ * D_;
  cvt_f32_bf16<<<WN / 256, 256, 0, stream>>>(Wq, wqb, WN);
  cvt_f32_bf16<<<WN / 256, 256, 0, stream>>>(Wk, wkb, WN);
  cvt_f32_bf16<<<WN / 256, 256, 0, stream>>>(Wv, wvb, WN);
  cvt_f32_bf16<<<WN / 256, 256, 0, stream>>>(Wo, wob, WN);

  dim3 pgrid((B_ * S_) / 64, D_ / 64);  // (128, 8)
  proj_qkv<<<pgrid, 128, 0, stream>>>(q, wqb, qb);
  proj_qkv<<<pgrid, 128, 0, stream>>>(k, wkb, kb);
  proj_qkv<<<pgrid, 128, 0, stream>>>(v, wvb, vb);

  attn_fwd<<<dim3(S_ / 64, B_ * H_), 128, 0, stream>>>(qb, kb, vb, ctxb);

  out_proj<<<pgrid, 128, 0, stream>>>(ctxb, wob, bo, out);
}
